// GPT2_20822001451498
// MI455X (gfx1250) — matrix operand order repaired and checked by reading, not compiled, not run on hardware
//
#include <hip/hip_runtime.h>
#include <hip/hip_bf16.h>

// ---------------------------------------------------------------------------
// GPT-2 (124M) forward for gfx1250 (MI455X).
// Matmuls: v_wmma_f32_16x16x32_f16 (f16 operands, f32 accumulate).
// GEMM tiles staged with GLOBAL_LOAD_ASYNC_TO_LDS (ASYNCcnt-tracked, double
// buffered) so HBM/L2 -> LDS traffic bypasses VGPRs and overlaps the WMMAs.
// Residual stream kept fp32; attention is flash-style (no TxT materialization).
// ---------------------------------------------------------------------------

typedef __attribute__((ext_vector_type(16))) _Float16 v16h;
typedef __attribute__((ext_vector_type(8)))  _Float16 h8;
typedef __attribute__((ext_vector_type(8)))  float    v8f;

#define GPT_B   2
#define GPT_T   1024
#define GPT_C   768
#define GPT_3C  2304
#define GPT_FF  3072
#define GPT_H   12
#define GPT_D   64
#define GPT_L   12
#define GPT_V   50257
#define N_TOK   (GPT_B * GPT_T)      // 2048
#define LOG2E   1.4426950408889634f

// ---- fragment load: v16h from two contiguous 16B chunks ----
static __device__ inline v16h ldfrag(const _Float16* p0, const _Float16* p1) {
    union { v16h v; h8 h[2]; } u;
    u.h[0] = *(const h8*)p0;
    u.h[1] = *(const h8*)p1;
    return u.v;
}

static __device__ inline v8f wmma_f16(v16h a, v16h b, v8f c) {
    return __builtin_amdgcn_wmma_f32_16x16x32_f16(
        /*neg_a=*/false, a, /*neg_b=*/false, b,
        /*c_mod=*/(short)0, c, /*reuse_a=*/false, /*reuse_b=*/false);
}

// ---- async global->LDS copy (16B per lane), tracked by ASYNCcnt ----
static __device__ inline void async_copy_b128(const _Float16* gsrc, _Float16* ldst) {
    unsigned int loff =
        (unsigned int)(__SIZE_TYPE__)(__attribute__((address_space(3))) _Float16*)ldst;
    asm volatile("global_load_async_to_lds_b128 %0, %1, off"
                 :: "v"(loff), "v"(gsrc) : "memory");
}
static __device__ inline void wait_async3() {
    asm volatile("s_wait_asynccnt 0x3" ::: "memory");
}
static __device__ inline void wait_async0() {
    asm volatile("s_wait_asynccnt 0x0" ::: "memory");
}

static __device__ inline float gelu_tanh(float x) {
    float x3 = x * x * x;
    return 0.5f * x * (1.0f + tanhf(0.7978845608028654f * (x + 0.044715f * x3)));
}

// ---------------------------------------------------------------------------
// fp32 -> f16 conversion (weights)
// ---------------------------------------------------------------------------
__global__ __launch_bounds__(256) void cvt_f32_to_f16(const float* __restrict__ in,
                                                      _Float16* __restrict__ out, int n) {
    int i = blockIdx.x * 256 + threadIdx.x;
    if (i < n) out[i] = (_Float16)in[i];
}

// ---------------------------------------------------------------------------
// Embedding: x[row, c] = tok_embed[tokens[row], c] + pos_embed[row % T, c]
// ---------------------------------------------------------------------------
__global__ __launch_bounds__(256) void embed_kernel(const int* __restrict__ tokens,
                                                    const float* __restrict__ te,
                                                    const float* __restrict__ pe,
                                                    float* __restrict__ x) {
    int row = blockIdx.x;
    int t = threadIdx.x;
    int tok = tokens[row];
    int pos = row & (GPT_T - 1);
#pragma unroll
    for (int i = 0; i < 3; i++) {
        int c = t + i * 256;
        x[(size_t)row * GPT_C + c] = te[(size_t)tok * GPT_C + c] + pe[(size_t)pos * GPT_C + c];
    }
}

// ---------------------------------------------------------------------------
// LayerNorm (C=768) -> f16 output. One block (256 thr) per row.
// ---------------------------------------------------------------------------
static __device__ float block_sum256(float v, float* sred) {
    int t = threadIdx.x;
#pragma unroll
    for (int off = 16; off > 0; off >>= 1) v += __shfl_xor(v, off, 32);
    if ((t & 31) == 0) sred[t >> 5] = v;
    __syncthreads();
    float r = 0.f;
#pragma unroll
    for (int i = 0; i < 8; i++) r += sred[i];
    __syncthreads();
    return r;
}

__global__ __launch_bounds__(256) void ln_kernel(const float* __restrict__ x,
                                                 const float* __restrict__ w,
                                                 const float* __restrict__ bb,
                                                 _Float16* __restrict__ out) {
    __shared__ float sred[8];
    int row = blockIdx.x, t = threadIdx.x;
    const float* xr = x + (size_t)row * GPT_C;
    float v[3];
#pragma unroll
    for (int i = 0; i < 3; i++) v[i] = xr[t + i * 256];
    float mean = block_sum256(v[0] + v[1] + v[2], sred) * (1.0f / GPT_C);
    float d0 = v[0] - mean, d1 = v[1] - mean, d2 = v[2] - mean;
    float var = block_sum256(d0 * d0 + d1 * d1 + d2 * d2, sred) * (1.0f / GPT_C);
    float rstd = rsqrtf(var + 1e-5f);
#pragma unroll
    for (int i = 0; i < 3; i++) {
        int c = t + i * 256;
        out[(size_t)row * GPT_C + c] = (_Float16)((v[i] - mean) * rstd * w[c] + bb[c]);
    }
}

// ---------------------------------------------------------------------------
// WMMA GEMM:  out[M,N] = A[M,K] * B[N,K]^T (+ bias) (+epilogue)
// Block tile 64x128, BK=32, 256 threads = 8 waves, each wave 32x32 output.
// Double-buffered async-to-LDS staging: while buffer b is consumed by WMMAs,
// buffer b^1 is filled by GLOBAL_LOAD_ASYNC_TO_LDS_B128 (3 per wave).
// EPI: 0 = f16 out, 1 = f32 out + residual add, 2 = GELU f16 out, 3 = f32 out
// ---------------------------------------------------------------------------
#define EPI_F16   0
#define EPI_RESID 1
#define EPI_GELU  2
#define EPI_F32   3

template <int EPI>
__global__ __launch_bounds__(256) void gemm_kernel(const _Float16* __restrict__ A,
                                                   const _Float16* __restrict__ B,
                                                   const float* __restrict__ bias,
                                                   const float* resid,
                                                   void* outp,
                                                   int M, int N, int K) {
    __shared__ _Float16 As[2][64][40];    // 32 + 8 pad halfs per row
    __shared__ _Float16 Bs[2][128][40];

    const int t = threadIdx.x;
    const int lane = t & 31, wv = t >> 5;
    const int hi = lane >> 4, lo = lane & 15;
    const int wm = wv >> 2, wn = wv & 3;         // 2 x 4 wave grid
    const int m0 = blockIdx.y * 64, n0 = blockIdx.x * 128;

    v8f acc[2][2];
#pragma unroll
    for (int mt = 0; mt < 2; mt++)
#pragma unroll
        for (int nt = 0; nt < 2; nt++)
#pragma unroll
            for (int e = 0; e < 8; e++) acc[mt][nt][e] = 0.0f;

    const int arow = t >> 2, apart = t & 3;
    const int brow0 = t >> 2, bpart0 = t & 3;          // chunk 0
    const int brow1 = (t + 256) >> 2, bpart1 = t & 3;  // chunk 1
    int nb0 = n0 + brow0; if (nb0 >= N) nb0 = N - 1;
    int nb1 = n0 + brow1; if (nb1 >= N) nb1 = N - 1;

    // stage(buf, kk): 3 async b128 copies per lane (1 A chunk + 2 B chunks)
#define GEMM_STAGE(buf, kk)                                                        \
    do {                                                                           \
        async_copy_b128(&A[(size_t)(m0 + arow) * K + (kk) + apart * 8],            \
                        &As[(buf)][arow][apart * 8]);                              \
        async_copy_b128(&B[(size_t)nb0 * K + (kk) + bpart0 * 8],                   \
                        &Bs[(buf)][brow0][bpart0 * 8]);                            \
        async_copy_b128(&B[(size_t)nb1 * K + (kk) + bpart1 * 8],                   \
                        &Bs[(buf)][brow1][bpart1 * 8]);                            \
    } while (0)

    GEMM_STAGE(0, 0);

    const int nsteps = K >> 5;
    for (int s = 0; s < nsteps; s++) {
        const int buf = s & 1;
        __syncthreads();                       // everyone done reading buf^1
        const bool pre = (s + 1) < nsteps;
        if (pre) GEMM_STAGE(buf ^ 1, (s + 1) << 5);
        if (pre) wait_async3();                // buf's 3 loads retired
        else     wait_async0();
        __syncthreads();                       // buf visible to all waves

        v16h af[2], bf[2];
#pragma unroll
        for (int mt = 0; mt < 2; mt++) {
            const _Float16* ar = &As[buf][wm * 32 + mt * 16 + lo][0];
            af[mt] = ldfrag(ar + 8 * hi, ar + 16 + 8 * hi);
        }
#pragma unroll
        for (int nt = 0; nt < 2; nt++) {
            const _Float16* br = &Bs[buf][wn * 32 + nt * 16 + lo][0];
            bf[nt] = ldfrag(br + 8 * hi, br + 16 + 8 * hi);
        }
#pragma unroll
        for (int mt = 0; mt < 2; mt++)
#pragma unroll
            for (int nt = 0; nt < 2; nt++)
                acc[mt][nt] = wmma_f16(af[mt], bf[nt], acc[mt][nt]);
    }
#undef GEMM_STAGE

    // epilogue
#pragma unroll
    for (int mt = 0; mt < 2; mt++) {
#pragma unroll
        for (int nt = 0; nt < 2; nt++) {
            int col = n0 + wn * 32 + nt * 16 + lo;
            float bv = (col < N && bias) ? bias[col] : 0.0f;
#pragma unroll
            for (int r = 0; r < 8; r++) {
                int row = m0 + wm * 32 + mt * 16 + r + 8 * hi;
                if (col < N) {
                    float v = acc[mt][nt][r] + bv;
                    size_t o = (size_t)row * N + col;
                    if (EPI == EPI_F16) {
                        ((_Float16*)outp)[o] = (_Float16)v;
                    } else if (EPI == EPI_RESID) {
                        ((float*)outp)[o] = resid[o] + v;
                    } else if (EPI == EPI_GELU) {
                        ((_Float16*)outp)[o] = (_Float16)gelu_tanh(v);
                    } else {
                        ((float*)outp)[o] = v;
                    }
                }
            }
        }
    }
}

// ---------------------------------------------------------------------------
// Flash attention. qkv: [N_TOK, 3C] f16 (q|k|v). o16: [N_TOK, C] f16.
// Grid: B*H*(T/128) workgroups, 256 threads = 8 waves, 16 q rows per wave.
// K-blocks of 32 keys staged in LDS; online softmax; WMMA for S and P*V.
// ---------------------------------------------------------------------------
__global__ __launch_bounds__(256) void attn_kernel(const _Float16* __restrict__ qkv,
                                                   _Float16* __restrict__ o16) {
    __shared__ _Float16 Ks[32][72];        // [key][d], pad to 72
    __shared__ _Float16 Vts[64][40];       // [d][key], pad to 40
    __shared__ _Float16 Pb[8][16][40];     // per-wave P staging

    const int t = threadIdx.x, lane = t & 31, wv = t >> 5;
    const int hi = lane >> 4, lo = lane & 15;

    int wg = blockIdx.x;
    int b = wg / (GPT_H * 8);
    int rem = wg % (GPT_H * 8);
    int h = rem / 8;
    int q0 = (rem % 8) * 128;
    const int rowbase = b * GPT_T;

    // Preload Q fragments for this wave's 16 q rows (k chunks d=0..31, 32..63)
    int qrow = rowbase + q0 + wv * 16 + lo;
    const _Float16* qp = qkv + (size_t)qrow * GPT_3C + h * GPT_D;
    v16h qf[2];
#pragma unroll
    for (int dc = 0; dc < 2; dc++)
        qf[dc] = ldfrag(qp + dc * 32 + 8 * hi, qp + dc * 32 + 16 + 8 * hi);

    float rm[8], rs[8];
    v8f accO[4];
#pragma unroll
    for (int r = 0; r < 8; r++) { rm[r] = -1e30f; rs[r] = 0.0f; }
#pragma unroll
    for (int dt = 0; dt < 4; dt++)
#pragma unroll
        for (int e = 0; e < 8; e++) accO[dt][e] = 0.0f;

    const int kmax = q0 + 128;               // causal bound for this block
    const int qlast = q0 + wv * 16 + 15;     // last q row this wave owns
    const int srow = t >> 3, spart = t & 7;  // staging mapping

    for (int kb = 0; kb < kmax; kb += 32) {
        // stage K (row-major) and V (transposed) for 32 keys
        const _Float16* kp = qkv + (size_t)(rowbase + kb + srow) * GPT_3C + GPT_C + h * GPT_D + spart * 8;
        *(h8*)&Ks[srow][spart * 8] = *(const h8*)kp;
        const _Float16* vp = qkv + (size_t)(rowbase + kb + srow) * GPT_3C + 2 * GPT_C + h * GPT_D + spart * 8;
        h8 vvec = *(const h8*)vp;
#pragma unroll
        for (int j = 0; j < 8; j++) Vts[spart * 8 + j][srow] = vvec[j];
        __syncthreads();

        if (kb <= qlast) {                   // wave-uniform: EXEC stays full
            // S = Q * K^T for 16 rows x 32 keys (two 16-wide tiles)
            v8f sacc[2];
#pragma unroll
            for (int nt = 0; nt < 2; nt++)
#pragma unroll
                for (int e = 0; e < 8; e++) sacc[nt][e] = 0.0f;
#pragma unroll
            for (int nt = 0; nt < 2; nt++) {
#pragma unroll
                for (int dc = 0; dc < 2; dc++) {
                    const _Float16* kr = &Ks[nt * 16 + lo][dc * 32 + 8 * hi];
                    v16h bk = ldfrag(kr, kr + 16);
                    sacc[nt] = wmma_f16(qf[dc], bk, sacc[nt]);
                }
            }
            // online softmax (rows live in 16-lane half-wave groups)
            float p0v[8], p1v[8];
#pragma unroll
            for (int r = 0; r < 8; r++) {
                int qr = q0 + wv * 16 + r + 8 * hi;
                float s0 = sacc[0][r] * 0.125f;
                float s1 = sacc[1][r] * 0.125f;
                if (kb + lo > qr)      s0 = -1e30f;
                if (kb + 16 + lo > qr) s1 = -1e30f;
                float m = fmaxf(s0, s1);
                m = fmaxf(m, __shfl_xor(m, 1, 32));
                m = fmaxf(m, __shfl_xor(m, 2, 32));
                m = fmaxf(m, __shfl_xor(m, 4, 32));
                m = fmaxf(m, __shfl_xor(m, 8, 32));
                float nm = fmaxf(rm[r], m);
                float corr = exp2f((rm[r] - nm) * LOG2E);
                float p0 = exp2f((s0 - nm) * LOG2E);
                float p1 = exp2f((s1 - nm) * LOG2E);
                float ts = p0 + p1;
                ts += __shfl_xor(ts, 1, 32);
                ts += __shfl_xor(ts, 2, 32);
                ts += __shfl_xor(ts, 4, 32);
                ts += __shfl_xor(ts, 8, 32);
                rs[r] = rs[r] * corr + ts;
                rm[r] = nm;
#pragma unroll
                for (int dt = 0; dt < 4; dt++) accO[dt][r] *= corr;
                p0v[r] = p0;
                p1v[r] = p1;
            }
            // route P through LDS: C-layout -> A-layout (wave-private, DS in-order)
#pragma unroll
            for (int r = 0; r < 8; r++) {
                Pb[wv][r + 8 * hi][lo]      = (_Float16)p0v[r];
                Pb[wv][r + 8 * hi][16 + lo] = (_Float16)p1v[r];
            }
            v16h pf = ldfrag(&Pb[wv][lo][8 * hi], &Pb[wv][lo][16 + 8 * hi]);
            // O += P * V  (4 d-tiles of 16)
#pragma unroll
            for (int dt = 0; dt < 4; dt++) {
                const _Float16* vr = &Vts[dt * 16 + lo][8 * hi];
                v16h bv = ldfrag(vr, vr + 16);
                accO[dt] = wmma_f16(pf, bv, accO[dt]);
            }
        }
        __syncthreads();
    }

    // write O (f16), normalized by row sums
#pragma unroll
    for (int dt = 0; dt < 4; dt++) {
#pragma unroll
        for (int r = 0; r < 8; r++) {
            int row = rowbase + q0 + wv * 16 + r + 8 * hi;
            o16[(size_t)row * GPT_C + h * GPT_D + dt * 16 + lo] =
                (_Float16)(accO[dt][r] * (1.0f / rs[r]));
        }
    }
}

// ---------------------------------------------------------------------------
// Host launcher
// ---------------------------------------------------------------------------
extern "C" void kernel_launch(void* const* d_in, const int* in_sizes, int n_in,
                              void* d_out, int out_size, void* d_ws, size_t ws_size,
                              hipStream_t stream) {
    const int*   tokens    = (const int*)d_in[0];
    const float* tok_embed = (const float*)d_in[1];
    const float* pos_embed = (const float*)d_in[2];
    const float* ln1_w     = (const float*)d_in[3];
    const float* ln1_b     = (const float*)d_in[4];
    const float* qkv_w     = (const float*)d_in[5];
    const float* qkv_b     = (const float*)d_in[6];
    const float* proj_w    = (const float*)d_in[7];
    const float* proj_b    = (const float*)d_in[8];
    const float* ln2_w     = (const float*)d_in[9];
    const float* ln2_b     = (const float*)d_in[10];
    const float* fc_w      = (const float*)d_in[11];
    const float* fc_b      = (const float*)d_in[12];
    const float* fc2_w     = (const float*)d_in[13];
    const float* fc2_b     = (const float*)d_in[14];
    const float* lnf_w     = (const float*)d_in[15];
    const float* lnf_b     = (const float*)d_in[16];
    float* logits = (float*)d_out;

    // workspace carve-up (256B aligned)
    size_t off = 0;
    auto alloc = [&](size_t bytes) -> char* {
        char* p = (char*)d_ws + off;
        off = (off + bytes + 255) & ~(size_t)255;
        return p;
    };
    const size_t n_wq  = (size_t)GPT_L * GPT_3C * GPT_C;
    const size_t n_wp  = (size_t)GPT_L * GPT_C * GPT_C;
    const size_t n_wf  = (size_t)GPT_L * GPT_FF * GPT_C;
    const size_t n_wf2 = (size_t)GPT_L * GPT_C * GPT_FF;
    const size_t n_emb = (size_t)GPT_V * GPT_C;

    _Float16* wq16  = (_Float16*)alloc(n_wq  * 2);
    _Float16* wp16  = (_Float16*)alloc(n_wp  * 2);
    _Float16* wf16  = (_Float16*)alloc(n_wf  * 2);
    _Float16* wf216 = (_Float16*)alloc(n_wf2 * 2);
    _Float16* emb16 = (_Float16*)alloc(n_emb * 2);
    float*    x     = (float*)   alloc((size_t)N_TOK * GPT_C * 4);
    _Float16* h16   = (_Float16*)alloc((size_t)N_TOK * GPT_C * 2);
    _Float16* qkv16 = (_Float16*)alloc((size_t)N_TOK * GPT_3C * 2);
    _Float16* o16   = (_Float16*)alloc((size_t)N_TOK * GPT_C * 2);
    _Float16* g16   = (_Float16*)alloc((size_t)N_TOK * GPT_FF * 2);
    (void)ws_size; (void)in_sizes; (void)n_in; (void)out_size;

    // weight conversions (fp32 -> f16)
    auto cvt = [&](const float* src, _Float16* dst, size_t n) {
        cvt_f32_to_f16<<<(unsigned)((n + 255) / 256), 256, 0, stream>>>(src, dst, (int)n);
    };
    cvt(qkv_w, wq16, n_wq);
    cvt(proj_w, wp16, n_wp);
    cvt(fc_w, wf16, n_wf);
    cvt(fc2_w, wf216, n_wf2);
    cvt(tok_embed, emb16, n_emb);

    // embeddings
    embed_kernel<<<N_TOK, 256, 0, stream>>>(tokens, tok_embed, pos_embed, x);

    const dim3 blk(256);
    for (int l = 0; l < GPT_L; l++) {
        // ln1 -> h16
        ln_kernel<<<N_TOK, blk, 0, stream>>>(x, ln1_w + l * GPT_C, ln1_b + l * GPT_C, h16);
        // qkv = h @ qw^T + qb  (f16 out)
        gemm_kernel<EPI_F16><<<dim3(GPT_3C / 128, N_TOK / 64), blk, 0, stream>>>(
            h16, wq16 + (size_t)l * GPT_3C * GPT_C, qkv_b + (size_t)l * GPT_3C,
            nullptr, qkv16, N_TOK, GPT_3C, GPT_C);
        // attention
        attn_kernel<<<GPT_B * GPT_H * (GPT_T / 128), blk, 0, stream>>>(qkv16, o16);
        // x = x + o @ pw^T + pb
        gemm_kernel<EPI_RESID><<<dim3(GPT_C / 128, N_TOK / 64), blk, 0, stream>>>(
            o16, wp16 + (size_t)l * GPT_C * GPT_C, proj_b + (size_t)l * GPT_C,
            x, x, N_TOK, GPT_C, GPT_C);
        // ln2 -> h16
        ln_kernel<<<N_TOK, blk, 0, stream>>>(x, ln2_w + l * GPT_C, ln2_b + l * GPT_C, h16);
        // g = gelu(h @ fw^T + fb)  (f16 out)
        gemm_kernel<EPI_GELU><<<dim3(GPT_FF / 128, N_TOK / 64), blk, 0, stream>>>(
            h16, wf16 + (size_t)l * GPT_FF * GPT_C, fc_b + (size_t)l * GPT_FF,
            nullptr, g16, N_TOK, GPT_FF, GPT_C);
        // x = x + g @ f2w^T + f2b
        gemm_kernel<EPI_RESID><<<dim3(GPT_C / 128, N_TOK / 64), blk, 0, stream>>>(
            g16, wf216 + (size_t)l * GPT_C * GPT_FF, fc2_b + (size_t)l * GPT_C,
            x, x, N_TOK, GPT_C, GPT_FF);
    }

    // final LN + tied LM head
    ln_kernel<<<N_TOK, blk, 0, stream>>>(x, lnf_w, lnf_b, h16);
    gemm_kernel<EPI_F32><<<dim3((GPT_V + 127) / 128, N_TOK / 64), blk, 0, stream>>>(
        h16, emb16, nullptr, nullptr, logits, N_TOK, GPT_V, GPT_C);
}
